// MMNerModel_49924699849347
// MI455X (gfx1250) — matrix operand order repaired and checked by reading, not compiled, not run on hardware
//
#include <hip/hip_runtime.h>
#include <math.h>

// ---------------- model dims ----------------
#define B_    32
#define L_    256
#define NOBJ  4
#define D_    512
#define H_    8
#define DK_   64
#define HID_  256
#define NL_   6
#define NT_   13
#define EPSF  1e-6f

// ---------------- WMMA types ----------------
typedef __attribute__((ext_vector_type(16))) __bf16 v16bf;
typedef __attribute__((ext_vector_type(8)))  float  v8f;

// native f32->bf16 (RNE on gfx1250; clang>=17 treats __bf16 as arithmetic type)
static __device__ __forceinline__ unsigned pack2bf(float a, float b) {
  union { __bf16 h[2]; unsigned u; } r;
  r.h[0] = (__bf16)a;
  r.h[1] = (__bf16)b;
  return r.u;
}

// =====================================================================
// Tiled bf16 WMMA GEMM:  C = scale * (A @ B) [+ bias] [relu]
// Block: 256 threads (8 wave32); block tile 128x64, K-step 32.
// Wave (4x2 grid) owns a 32x32 output tile: 2 A-frags x 2 B-frags -> 4 WMMA.
// LDS tiles are double-buffered (ping-pong) in WMMA fragment-major layout
// [subtile][lane][16 bf16], so each operand is one contiguous 32B LDS read
// (2x ds_load_b128); one barrier per K-step, next tile staged while the
// current one feeds the matrix pipe.
// Batch offset: off(z) = (z / hd) * s + (z % hd) * h   (hd=1 => z*s)
// TRANSB: B logical [K,N] stored as [N,K] row-major (for Q @ K^T).
// =====================================================================
#define AREC (8 * 32 * 16)   // A buffer: 8 M-subtiles
#define BREC (4 * 32 * 16)   // B buffer: 4 N-subtiles

template<bool TRANSB>
__global__ void __launch_bounds__(256) gemm_wmma(
    const float* __restrict__ A, const float* __restrict__ Bm,
    const float* __restrict__ bias, float* __restrict__ C,
    int M, int N, int K, int lda, int ldb, int ldc,
    long long sA, long long hA, int dA,
    long long sB, long long hB, int dB,
    long long sC, long long hC, int dC,
    float scale, int relu)
{
  __shared__ __attribute__((aligned(16))) __bf16 AsF[2 * AREC];  // 16KB
  __shared__ __attribute__((aligned(16))) __bf16 BsF[2 * BREC];  //  8KB

  const int z = blockIdx.z;
  const float* Ap = A  + (long long)(z / dA) * sA + (long long)(z % dA) * hA;
  const float* Bp = Bm + (long long)(z / dB) * sB + (long long)(z % dB) * hB;
  float*       Cp = C  + (long long)(z / dC) * sC + (long long)(z % dC) * hC;

  const int tileM = blockIdx.y * 128;
  const int tileN = blockIdx.x * 64;
  const int tid   = threadIdx.x;
  const int lane  = tid & 31;
  const int wave  = tid >> 5;
  const int wm    = wave >> 1;        // 0..3 -> rows wm*32
  const int wn    = wave & 1;         // 0..1 -> cols wn*32
  const int lr    = lane & 15;
  const int hi    = lane >> 4;        // 0 | 1

  v8f acc00 = {}, acc01 = {}, acc10 = {}, acc11 = {};

  // ---- staging geometry (fixed across K loop) ----
  // A: thread reads row ar, 16 consecutive fp32 (ch..ch+15) -> two record halves
  //    record[sub= r>>4][ (r&15) + 16*(c>>4) ][ c&15 ]
  const int ar = tid >> 1;            // 0..127
  const int ch = (tid & 1) * 16;      // 0 | 16
  const int aoff0 = ((ar >> 4) * 32 + (ar & 15) + (ch >> 4) * 16) * 16;        // c=ch..ch+7
  const int aoff1 = ((ar >> 4) * 32 + (ar & 15) + (ch >> 4) * 16) * 16 + 8;    // c=ch+8..ch+15

  auto stage = [&](int kk0, int pb) {
    // ---- A: 128x32 tile, 16 fp32 per thread -> two 16B LDS stores ----
    {
      const float* src = Ap + (long long)(tileM + ar) * lda + (kk0 + ch);
      uint4 p0, p1;
      p0.x = pack2bf(src[0],  src[1]);  p0.y = pack2bf(src[2],  src[3]);
      p0.z = pack2bf(src[4],  src[5]);  p0.w = pack2bf(src[6],  src[7]);
      p1.x = pack2bf(src[8],  src[9]);  p1.y = pack2bf(src[10], src[11]);
      p1.z = pack2bf(src[12], src[13]); p1.w = pack2bf(src[14], src[15]);
      __bf16* base = AsF + pb * AREC;
      *(uint4*)(base + aoff0) = p0;
      *(uint4*)(base + aoff1) = p1;
      if (kk0 + 32 < K) __builtin_prefetch(src + 32, 0, 1);
    }
    // ---- B: 32x64 tile into record[nsub][(c&15)+16*(k>=16)][k&15] ----
    if (TRANSB) {
      // 8 consecutive K of one column of K^T (coalesced) -> one record half
      const int c  = tid >> 2;            // 0..63
      const int kk = (tid & 3) * 8;       // 0,8,16,24
      const int gn = tileN + c;
      float f0=0,f1=0,f2=0,f3=0,f4=0,f5=0,f6=0,f7=0;
      if (gn < N) {
        const float* src = Bp + (long long)gn * ldb + (kk0 + kk);
        f0=src[0]; f1=src[1]; f2=src[2]; f3=src[3];
        f4=src[4]; f5=src[5]; f6=src[6]; f7=src[7];
        if (kk0 + 32 < K) __builtin_prefetch(src + 32, 0, 1);
      }
      uint4 p;
      p.x = pack2bf(f0, f1); p.y = pack2bf(f2, f3);
      p.z = pack2bf(f4, f5); p.w = pack2bf(f6, f7);
      __bf16* bdst = BsF + pb * BREC
                   + ((((c >> 4) * 32) + (c & 15) + ((kk >> 4) * 16)) * 16 + (kk & 15));
      *(uint4*)bdst = p;
    } else {
      // 8 consecutive N of one K row (coalesced); scatter 8 u16 into records
      const int kb  = tid >> 3;           // 0..31
      const int c0b = (tid & 7) * 8;      // 0..56
      const float* src = Bp + (long long)(kk0 + kb) * ldb + (tileN + c0b);
      if (kk0 + 32 < K && (tileN + c0b) < N) __builtin_prefetch(src + (long long)32 * ldb, 0, 1);
      __bf16* base = BsF + pb * BREC;
      #pragma unroll
      for (int j = 0; j < 8; ++j) {
        const int c  = c0b + j;
        const int gn = tileN + c;
        const float v = (gn < N) ? src[j] : 0.f;
        base[(((c >> 4) * 32) + (c & 15) + ((kb >> 4) * 16)) * 16 + (kb & 15)] = (__bf16)v;
      }
    }
  };

  stage(0, 0);
  __syncthreads();

  int p = 0;
  for (int k0 = 0; k0 < K; k0 += 32) {
    // ---- wide fragment loads (32B contiguous per lane) ----
    const v16bf* Av = (const v16bf*)(AsF + p * AREC);
    const v16bf* Bv = (const v16bf*)(BsF + p * BREC);
    v16bf a0 = Av[(wm * 2 + 0) * 32 + lane];
    v16bf a1 = Av[(wm * 2 + 1) * 32 + lane];
    v16bf b0 = Bv[(wn * 2 + 0) * 32 + lane];
    v16bf b1 = Bv[(wn * 2 + 1) * 32 + lane];

    // ---- stage next K tile into the other buffer while WMMAs run ----
    if (k0 + 32 < K) stage(k0 + 32, p ^ 1);

    acc00 = __builtin_amdgcn_wmma_f32_16x16x32_bf16(false, a0, false, b0, (short)0, acc00, false, false);
    acc01 = __builtin_amdgcn_wmma_f32_16x16x32_bf16(false, a0, false, b1, (short)0, acc01, false, false);
    acc10 = __builtin_amdgcn_wmma_f32_16x16x32_bf16(false, a1, false, b0, (short)0, acc10, false, false);
    acc11 = __builtin_amdgcn_wmma_f32_16x16x32_bf16(false, a1, false, b1, (short)0, acc11, false, false);

    __syncthreads();
    p ^= 1;
  }

  // ---- epilogue: C lane map -> row = vr + 8*hi, col = lane&15 ----
  const int colBase  = tileN + wn * 32 + lr;
  const int rowBase0 = tileM + wm * 32 + hi * 8;       // a0 tiles
  const int rowBase1 = rowBase0 + 16;                  // a1 tiles
  #pragma unroll
  for (int vr = 0; vr < 8; ++vr) {
    const int r0 = rowBase0 + vr;
    const int r1 = rowBase1 + vr;
    int col = colBase;
    if (col < N) {
      float v = acc00[vr] * scale;
      if (bias) v += bias[col];
      if (relu) v = fmaxf(v, 0.f);
      Cp[(long long)r0 * ldc + col] = v;
      v = acc10[vr] * scale;
      if (bias) v += bias[col];
      if (relu) v = fmaxf(v, 0.f);
      Cp[(long long)r1 * ldc + col] = v;
    }
    col = colBase + 16;
    if (col < N) {
      float v = acc01[vr] * scale;
      if (bias) v += bias[col];
      if (relu) v = fmaxf(v, 0.f);
      Cp[(long long)r0 * ldc + col] = v;
      v = acc11[vr] * scale;
      if (bias) v += bias[col];
      if (relu) v = fmaxf(v, 0.f);
      Cp[(long long)r1 * ldc + col] = v;
    }
  }
}

// =====================================================================
// LayerNorm (annotated-transformer): (x-mean)/(std_ddof1 + eps)*g + b
// One block per row, D==512, 2 elems/thread, optional fused residual r.
// =====================================================================
__global__ void __launch_bounds__(256) ln_kernel(
    const float* __restrict__ a, const float* __restrict__ r,
    const float* __restrict__ g, const float* __restrict__ be,
    float* __restrict__ out, int D)
{
  __shared__ float red[256];
  const long long row = blockIdx.x;
  const int tid = threadIdx.x;
  float v0 = a[row * D + tid];
  float v1 = a[row * D + tid + 256];
  if (r) { v0 += r[row * D + tid]; v1 += r[row * D + tid + 256]; }
  red[tid] = v0 + v1; __syncthreads();
  for (int off = 128; off; off >>= 1) { if (tid < off) red[tid] += red[tid + off]; __syncthreads(); }
  const float mean = red[0] / (float)D;
  __syncthreads();
  const float d0 = v0 - mean, d1 = v1 - mean;
  red[tid] = d0 * d0 + d1 * d1; __syncthreads();
  for (int off = 128; off; off >>= 1) { if (tid < off) red[tid] += red[tid + off]; __syncthreads(); }
  const float var = red[0] / (float)(D - 1);        // unbiased (ddof=1)
  const float inv = 1.f / (sqrtf(var) + EPSF);      // std + eps
  out[row * D + tid]       = g[tid] * d0 * inv + be[tid];
  out[row * D + tid + 256] = g[tid + 256] * d1 * inv + be[tid + 256];
}

// masked softmax over last dim (L==256==blockDim); scale pre-applied by GEMM
__global__ void __launch_bounds__(256) softmax_kernel(
    float* __restrict__ s, const int* __restrict__ mask, int L, int Hh)
{
  __shared__ float red[256];
  const long long row = blockIdx.x;
  const int tid = threadIdx.x;
  const int b = (int)((row / L) / Hh);
  float* p = s + row * (long long)L;
  float v = p[tid];
  if (mask && mask[b * L + tid] == 0) v = -1e9f;
  red[tid] = v; __syncthreads();
  for (int off = 128; off; off >>= 1) { if (tid < off) red[tid] = fmaxf(red[tid], red[tid + off]); __syncthreads(); }
  const float mx = red[0]; __syncthreads();
  const float e = expf(v - mx);
  red[tid] = e; __syncthreads();
  for (int off = 128; off; off >>= 1) { if (tid < off) red[tid] += red[tid + off]; __syncthreads(); }
  p[tid] = e / red[0];
}

// tiny object self-attention (N=4): V = o, no out-proj; block 64 = dk
__global__ void __launch_bounds__(64) oattn_kernel(
    const float* __restrict__ Q, const float* __restrict__ Km,
    const float* __restrict__ V, float* __restrict__ O)
{
  __shared__ float red[64];
  __shared__ float scs[NOBJ];
  const int d = threadIdx.x;
  const int idx = blockIdx.x;
  const int q = idx % NOBJ;
  const int h = (idx / NOBJ) % H_;
  const int b = idx / (NOBJ * H_);
  const int col = h * DK_ + d;
  const long long base = (long long)b * NOBJ * D_;
  const float qv = Q[base + (long long)q * D_ + col];
  for (int k = 0; k < NOBJ; ++k) {
    red[d] = qv * Km[base + (long long)k * D_ + col];
    __syncthreads();
    for (int off = 32; off; off >>= 1) { if (d < off) red[d] += red[d + off]; __syncthreads(); }
    if (d == 0) scs[k] = red[0] * 0.125f;   // 1/sqrt(64)
    __syncthreads();
  }
  if (d == 0) {
    float mx = scs[0];
    for (int k = 1; k < NOBJ; ++k) mx = fmaxf(mx, scs[k]);
    float sum = 0.f;
    for (int k = 0; k < NOBJ; ++k) { scs[k] = expf(scs[k] - mx); sum += scs[k]; }
    for (int k = 0; k < NOBJ; ++k) scs[k] /= sum;
  }
  __syncthreads();
  float outv = 0.f;
  for (int k = 0; k < NOBJ; ++k) outv += scs[k] * V[base + (long long)k * D_ + col];
  O[base + (long long)q * D_ + col] = outv;
}

// x2o[b,l,d] = sum_n sigmoid(gx[b,l,d] + go[b,n,d] + bg[d]) * inter[b,l,n] * newo[b,n,d]
__global__ void x2o_kernel(const float* __restrict__ gx, const float* __restrict__ go,
                           const float* __restrict__ bg, const float* __restrict__ inter,
                           const float* __restrict__ newo, float* __restrict__ out, int total)
{
  const int idx = blockIdx.x * blockDim.x + threadIdx.x;
  if (idx >= total) return;
  const int d = idx % D_;
  const int l = (idx / D_) % L_;
  const int b = idx / (D_ * L_);
  const float g0 = gx[idx] + bg[d];
  float acc = 0.f;
  #pragma unroll
  for (int n = 0; n < NOBJ; ++n) {
    const float z = g0 + go[((long long)b * NOBJ + n) * D_ + d];
    const float gt = 1.f / (1.f + expf(-z));
    acc += gt * inter[((long long)b * L_ + l) * NOBJ + n]
              * newo[((long long)b * NOBJ + n) * D_ + d];
  }
  out[idx] = acc;
}

// o2x[b,n,d] = sum_l sigmoid(gx[b,l,d] + go[b,n,d] + bg[d]) * inter[b,l,n] * newx[b,l,d]
__global__ void o2x_kernel(const float* __restrict__ gx, const float* __restrict__ go,
                           const float* __restrict__ bg, const float* __restrict__ inter,
                           const float* __restrict__ newx, float* __restrict__ out, int total)
{
  const int idx = blockIdx.x * blockDim.x + threadIdx.x;
  if (idx >= total) return;
  const int d = idx % D_;
  const int n = (idx / D_) % NOBJ;
  const int b = idx / (D_ * NOBJ);
  const float g1 = go[idx] + bg[d];
  float acc = 0.f;
  for (int l = 0; l < L_; ++l) {
    const float z = gx[((long long)b * L_ + l) * D_ + d] + g1;
    const float gt = 1.f / (1.f + expf(-z));
    acc += gt * inter[((long long)b * L_ + l) * NOBJ + n]
              * newx[((long long)b * L_ + l) * D_ + d];
  }
  out[idx] = acc;
}

__global__ void concat_kernel(const float* __restrict__ bx, const float* __restrict__ x,
                              float* __restrict__ out, int total)
{
  const int idx = blockIdx.x * blockDim.x + threadIdx.x;
  if (idx >= total) return;
  const int c = idx % (768 + D_);
  const long long row = idx / (768 + D_);
  out[idx] = (c < 768) ? bx[row * 768 + c] : x[row * D_ + (c - 768)];
}

// ---------------- host-side launch helpers ----------------
static void gemm(hipStream_t st, const float* A, const float* Bm, const float* bias, float* C,
                 int M, int N, int K, int lda, int ldb, int ldc, int batch,
                 long long sA, long long hA, int dA,
                 long long sB, long long hB, int dB,
                 long long sC, long long hC, int dC,
                 float scale, int relu, bool transB)
{
  dim3 g((unsigned)((N + 63) / 64), (unsigned)((M + 127) / 128), (unsigned)batch);
  dim3 b(256, 1, 1);
  if (transB)
    gemm_wmma<true ><<<g, b, 0, st>>>(A, Bm, bias, C, M, N, K, lda, ldb, ldc,
                                      sA, hA, dA, sB, hB, dB, sC, hC, dC, scale, relu);
  else
    gemm_wmma<false><<<g, b, 0, st>>>(A, Bm, bias, C, M, N, K, lda, ldb, ldc,
                                      sA, hA, dA, sB, hB, dB, sC, hC, dC, scale, relu);
}

static void gemm1(hipStream_t st, const float* A, const float* W, const float* bias, float* C,
                  int M, int N, int K, int relu = 0)
{
  gemm(st, A, W, bias, C, M, N, K, K, N, N, 1, 0, 0, 1, 0, 0, 1, 0, 0, 1, 1.f, relu, false);
}

extern "C" void kernel_launch(void* const* d_in, const int* in_sizes, int n_in,
                              void* d_out, int out_size, void* d_ws, size_t ws_size,
                              hipStream_t stream)
{
  (void)in_sizes; (void)n_in; (void)out_size; (void)ws_size;
  const float* bert_x  = (const float*)d_in[0];
  const float* o_feats = (const float*)d_in[1];
  const float* inter   = (const float*)d_in[2];
  const int*   tmask   = (const int*)d_in[3];
  const float* Wt  = (const float*)d_in[4];  const float* bt  = (const float*)d_in[5];
  const float* Wo1 = (const float*)d_in[6];  const float* bo1 = (const float*)d_in[7];
  const float* Wo2 = (const float*)d_in[8];  const float* bo2 = (const float*)d_in[9];
  const float* Wxq = (const float*)d_in[10]; const float* Wxk = (const float*)d_in[11];
  const float* Wxv = (const float*)d_in[12]; const float* Wxo = (const float*)d_in[13];
  const float* Woq = (const float*)d_in[14]; const float* Wok = (const float*)d_in[15];
  const float* Wgxo_x = (const float*)d_in[16]; const float* Wgxo_o = (const float*)d_in[17];
  const float* Wgox_x = (const float*)d_in[18]; const float* Wgox_o = (const float*)d_in[19];
  const float* bxq = (const float*)d_in[20]; const float* bxk = (const float*)d_in[21];
  const float* bxv = (const float*)d_in[22]; const float* bxo = (const float*)d_in[23];
  const float* boq = (const float*)d_in[24]; const float* bok = (const float*)d_in[25];
  const float* bgxo = (const float*)d_in[26]; const float* bgox = (const float*)d_in[27];
  const float* Wf1x = (const float*)d_in[28]; const float* bf1x = (const float*)d_in[29];
  const float* Wf2x = (const float*)d_in[30]; const float* bf2x = (const float*)d_in[31];
  const float* Wf1o = (const float*)d_in[32]; const float* bf1o = (const float*)d_in[33];
  const float* Wf2o = (const float*)d_in[34]; const float* bf2o = (const float*)d_in[35];
  const float* g_mesx = (const float*)d_in[36]; const float* g_meso = (const float*)d_in[37];
  const float* g_xg   = (const float*)d_in[38]; const float* g_og   = (const float*)d_in[39];
  const float* g_fx   = (const float*)d_in[40]; const float* g_fo   = (const float*)d_in[41];
  const float* b_mesx = (const float*)d_in[42]; const float* b_meso = (const float*)d_in[43];
  const float* b_xg   = (const float*)d_in[44]; const float* b_og   = (const float*)d_in[45];
  const float* b_fx   = (const float*)d_in[46]; const float* b_fo   = (const float*)d_in[47];
  const float* Wtag = (const float*)d_in[48]; const float* btag = (const float*)d_in[49];

  const int Mx = B_ * L_;      // 8192 text rows
  const int Mo = B_ * NOBJ;    // 128 object rows
  const int BH = B_ * H_;      // 256 attention batches
  const size_t SZX = (size_t)Mx * D_;
  const size_t SZO = (size_t)Mo * D_;
  const size_t SZS = (size_t)BH * L_ * L_;

  float* w = (float*)d_ws;
  float* xb   = w; w += SZX;        // x (layer carry)
  float* newx = w; w += SZX;
  float* t0   = w; w += SZX;        // Q / outproj / newx2
  float* t1   = w; w += SZX;        // K / gate-x / ffn out
  float* t2   = w; w += SZX;        // V / x2o
  float* t3   = w; w += SZX;        // ctx (merged via head-view store) / gate-x2
  float* sc   = w; w += SZS;        // scores; reused for FFN hidden and concat
  float* ob   = w; w += SZO;        // o (layer carry)
  float* u0   = w; w += SZO;
  float* u1   = w; w += SZO;
  float* u2   = w; w += SZO;
  float* u3   = w; w += SZO;        // newo

  // ---- input projections ----
  gemm1(stream, bert_x, Wt, bt, xb, Mx, D_, 768);
  gemm1(stream, o_feats, Wo1, bo1, u0, Mo, D_, 2048, 1);
  gemm1(stream, u0, Wo2, bo2, ob, Mo, D_, D_, 1);

  for (int i = 0; i < NL_; ++i) {
    const size_t wo = (size_t)i * D_ * D_;
    // ---- text self-attention ----
    gemm1(stream, xb, Wxq + wo, bxq + i * D_, t0, Mx, D_, D_);
    gemm1(stream, xb, Wxk + wo, bxk + i * D_, t1, Mx, D_, D_);
    gemm1(stream, xb, Wxv + wo, bxv + i * D_, t2, Mx, D_, D_);
    // scores = Q K^T / 8, reading [B,L,H*DK] in head-view (no transpose kernels)
    gemm(stream, t0, t1, nullptr, sc, L_, L_, DK_, D_, D_, L_, BH,
         (long long)L_ * D_, DK_, H_,
         (long long)L_ * D_, DK_, H_,
         (long long)L_ * L_, 0, 1, 0.125f, 0, true);
    softmax_kernel<<<BH * L_, 256, 0, stream>>>(sc, tmask, L_, H_);
    // ctx = A V, head-view store => merged [B,L,D] directly into t3
    gemm(stream, sc, t2, nullptr, t3, L_, DK_, L_, L_, D_, D_, BH,
         (long long)L_ * L_, 0, 1,
         (long long)L_ * D_, DK_, H_,
         (long long)L_ * D_, DK_, H_, 1.f, 0, false);
    gemm1(stream, t3, Wxo + wo, bxo + i * D_, t0, Mx, D_, D_);
    ln_kernel<<<Mx, 256, 0, stream>>>(xb, t0, g_mesx + i * D_, b_mesx + i * D_, newx, D_);

    // ---- object self-attention (V = o, no out-proj) ----
    gemm1(stream, ob, Woq + wo, boq + i * D_, u0, Mo, D_, D_);
    gemm1(stream, ob, Wok + wo, bok + i * D_, u1, Mo, D_, D_);
    oattn_kernel<<<B_ * H_ * NOBJ, 64, 0, stream>>>(u0, u1, ob, u2);
    ln_kernel<<<Mo, 256, 0, stream>>>(ob, u2, g_meso + i * D_, b_meso + i * D_, u3, D_); // u3 = newo

    // ---- gated x<-o messages ----
    gemm1(stream, newx, Wgxo_x + wo, nullptr, t1, Mx, D_, D_);
    gemm1(stream, u3,   Wgxo_o + wo, nullptr, u0, Mo, D_, D_);
    x2o_kernel<<<(int)((SZX + 255) / 256), 256, 0, stream>>>(t1, u0, bgxo + i * D_, inter, u3, t2, (int)SZX);
    // ---- gated o<-x messages ----
    gemm1(stream, newx, Wgox_x + wo, nullptr, t3, Mx, D_, D_);
    gemm1(stream, u3,   Wgox_o + wo, nullptr, u1, Mo, D_, D_);
    o2x_kernel<<<(int)((SZO + 255) / 256), 256, 0, stream>>>(t3, u1, bgox + i * D_, inter, newx, u2, (int)SZO);

    ln_kernel<<<Mx, 256, 0, stream>>>(newx, t2, g_xg + i * D_, b_xg + i * D_, t0, D_);  // t0 = newx2
    ln_kernel<<<Mo, 256, 0, stream>>>(u3,   u2, g_og + i * D_, b_og + i * D_, u0, D_);  // u0 = newo2

    // ---- FFNs + final LN of layer ----
    gemm1(stream, t0, Wf1x + (size_t)i * D_ * HID_, bf1x + i * HID_, sc, Mx, HID_, D_, 1);
    gemm1(stream, sc, Wf2x + (size_t)i * HID_ * D_, bf2x + i * D_,   t1, Mx, D_, HID_);
    ln_kernel<<<Mx, 256, 0, stream>>>(t0, t1, g_fx + i * D_, b_fx + i * D_, xb, D_);

    gemm1(stream, u0, Wf1o + (size_t)i * D_ * HID_, bf1o + i * HID_, u1, Mo, HID_, D_, 1);
    gemm1(stream, u1, Wf2o + (size_t)i * HID_ * D_, bf2o + i * D_,   u3, Mo, D_, HID_);
    ln_kernel<<<Mo, 256, 0, stream>>>(u0, u3, g_fo + i * D_, b_fo + i * D_, ob, D_);
  }

  // ---- concat(bert_x, x) @ Wtag + btag -> [B,L,13] ----
  concat_kernel<<<(Mx * (768 + D_) + 255) / 256, 256, 0, stream>>>(bert_x, xb, sc, Mx * (768 + D_));
  gemm1(stream, sc, Wtag, btag, (float*)d_out, Mx, NT_, 768 + D_);
}
